// GRU_1400159338668
// MI455X (gfx1250) — compile-verified, matrix operand order checked by reading, not run on hardware
//
#include <hip/hip_runtime.h>
#include <hip/hip_bf16.h>

typedef __attribute__((ext_vector_type(16))) __bf16 v16bf;
typedef __attribute__((ext_vector_type(8)))  __bf16 v8bf;
typedef __attribute__((ext_vector_type(4)))  __bf16 v4bf;
typedef __attribute__((ext_vector_type(8)))  float  v8f;

#define CAT16(lo, hi) __builtin_shufflevector(lo, hi, 0,1,2,3,4,5,6,7,8,9,10,11,12,13,14,15)

namespace gru_cfg {
constexpr int B  = 64;
constexpr int T  = 512;
constexpr int I  = 1024;
constexpr int H  = 1024;
constexpr int BT = B * T;      // 32768 GEMM rows
constexpr int G3 = 3 * H;      // 3072 gate columns
}

// ---------------------------------------------------------------------------
// fp32 -> bf16 weight conversion (runs once per launch; negligible traffic)
// ---------------------------------------------------------------------------
__global__ __launch_bounds__(256) void f32_to_bf16_kernel(const float* __restrict__ s,
                                                          __bf16* __restrict__ d, int n) {
  int i = blockIdx.x * 256 + threadIdx.x;
  if (i < n) d[i] = (__bf16)s[i];
}

__global__ __launch_bounds__(256) void zero_h_kernel(float* __restrict__ hf,
                                                     __bf16* __restrict__ hb, int n) {
  int i = blockIdx.x * 256 + threadIdx.x;
  if (i < n) { hf[i] = 0.0f; hb[i] = (__bf16)0.0f; }
}

// ---------------------------------------------------------------------------
// gi = x @ W_ih^T + b_ih      M=BT=32768, N=3H=3072, K=I=1024
// Block tile 128(M) x 64(N), 8 waves in 4x2, each wave register-blocked 2x2
// (four 16x16 accumulators) -> 4 v_wmma per 8 ds_load_b128 per K=32 slab.
// A (fp32 x) converted to bf16 through LDS; B (bf16 W_ih) staged via LDS.
// ---------------------------------------------------------------------------
__global__ __launch_bounds__(256) void gi_gemm_kernel(const float* __restrict__ x,
                                                      const __bf16* __restrict__ wih,
                                                      const float* __restrict__ bih,
                                                      float* __restrict__ gi) {
  using namespace gru_cfg;
  __shared__ __align__(16) __bf16 sA[128 * 32];  // 8 KB
  __shared__ __align__(16) __bf16 sB[64 * 32];   // 4 KB

  const int tid   = threadIdx.x;
  const int lane  = tid & 31;
  const int w     = tid >> 5;        // 0..7
  const int mw    = w >> 1;          // 0..3  (32-row slice within 128)
  const int nw    = w & 1;           // 0..1  (32-col slice within 64)
  const long mbase = (long)blockIdx.y * 128;
  const int  nbase = blockIdx.x * 64;

  const int r16   = lane & 15;
  const int khalf = lane >> 4;       // 0 or 1

  // staging: A 128x32 fp32 (16 per thread), B 64x32 bf16 (8 per thread)
  const int arow = tid >> 1;         // 0..127
  const int acol = (tid & 1) * 16;   // 0 or 16
  const int brow = tid >> 2;         // 0..63
  const int bcol = (tid & 3) * 8;    // 0,8,16,24

  v8f c00 = {}, c01 = {}, c10 = {}, c11 = {};

  for (int k0 = 0; k0 < I; k0 += 32) {
    // --- stage A: fp32 -> bf16 into LDS (v_cvt_pk_bf16_f32 path) ---
    {
      const float* src = x + (mbase + arow) * (long)I + k0 + acol;
      float4 f0 = *(const float4*)(src);
      float4 f1 = *(const float4*)(src + 4);
      float4 f2 = *(const float4*)(src + 8);
      float4 f3 = *(const float4*)(src + 12);
      v8bf va, vb;
      va[0] = (__bf16)f0.x; va[1] = (__bf16)f0.y; va[2] = (__bf16)f0.z; va[3] = (__bf16)f0.w;
      va[4] = (__bf16)f1.x; va[5] = (__bf16)f1.y; va[6] = (__bf16)f1.z; va[7] = (__bf16)f1.w;
      vb[0] = (__bf16)f2.x; vb[1] = (__bf16)f2.y; vb[2] = (__bf16)f2.z; vb[3] = (__bf16)f2.w;
      vb[4] = (__bf16)f3.x; vb[5] = (__bf16)f3.y; vb[6] = (__bf16)f3.z; vb[7] = (__bf16)f3.w;
      *(v8bf*)&sA[arow * 32 + acol]     = va;
      *(v8bf*)&sA[arow * 32 + acol + 8] = vb;
      if (k0 + 32 < I) __builtin_prefetch(src + 32, 0, 1);   // global_prefetch_b8
    }
    // --- stage B: bf16 weights into LDS ---
    {
      const __bf16* src = wih + (long)(nbase + brow) * I + k0 + bcol;
      *(v8bf*)&sB[brow * 32 + bcol] = *(const v8bf*)src;
    }
    __syncthreads();

    // --- fragments per 16-bit WMMA layout: two contiguous 8-elt K runs ---
    const int a0 = (mw * 32 + r16) * 32 + khalf * 8;        // mi = 0
    const int a1 = a0 + 16 * 32;                            // mi = 1
    v16bf fa0 = CAT16(*(const v8bf*)&sA[a0], *(const v8bf*)&sA[a0 + 16]);
    v16bf fa1 = CAT16(*(const v8bf*)&sA[a1], *(const v8bf*)&sA[a1 + 16]);
    const int b0 = (nw * 32 + r16) * 32 + khalf * 8;        // ni = 0
    const int b1 = b0 + 16 * 32;                            // ni = 1
    v16bf fb0 = CAT16(*(const v8bf*)&sB[b0], *(const v8bf*)&sB[b0 + 16]);
    v16bf fb1 = CAT16(*(const v8bf*)&sB[b1], *(const v8bf*)&sB[b1 + 16]);

    c00 = __builtin_amdgcn_wmma_f32_16x16x32_bf16(false, fa0, false, fb0, (short)0, c00, false, false);
    c01 = __builtin_amdgcn_wmma_f32_16x16x32_bf16(false, fa0, false, fb1, (short)0, c01, false, false);
    c10 = __builtin_amdgcn_wmma_f32_16x16x32_bf16(false, fa1, false, fb0, (short)0, c10, false, false);
    c11 = __builtin_amdgcn_wmma_f32_16x16x32_bf16(false, fa1, false, fb1, (short)0, c11, false, false);
    __syncthreads();
  }

  // epilogue: + bias, store 4 tiles (lane holds rows khalf*8..+7 of col r16)
  const int  col0 = nbase + nw * 32 + r16;
  const int  col1 = col0 + 16;
  const float bias0 = bih[col0];
  const float bias1 = bih[col1];
  const long row00 = mbase + mw * 32 + khalf * 8;
  const long row10 = row00 + 16;
#pragma unroll
  for (int i = 0; i < 8; ++i) {
    gi[(row00 + i) * (long)G3 + col0] = c00[i] + bias0;
    gi[(row00 + i) * (long)G3 + col1] = c01[i] + bias1;
    gi[(row10 + i) * (long)G3 + col0] = c10[i] + bias0;
    gi[(row10 + i) * (long)G3 + col1] = c11[i] + bias1;
  }
}

// ---------------------------------------------------------------------------
// One recurrence step (fused gh GEMM + gates).
// Grid: (H/16, 2).  Block: 6 waves = {gate 0..2} x {row-tile 0..1}, 192 thr.
// Each wave: 16x16 gh tile over K=1024 (32 WMMAs). Gates combined via LDS.
// W_hh (bf16, 6.3 MB) is L2-resident across all 512 steps.
// ---------------------------------------------------------------------------
__global__ __launch_bounds__(192) void gru_step_kernel(const float* __restrict__ gi,
                                                       const __bf16* __restrict__ whh,
                                                       const float* __restrict__ bhh,
                                                       const float* __restrict__ h_in,
                                                       const __bf16* __restrict__ h_in_bf,
                                                       float* __restrict__ h_out,
                                                       __bf16* __restrict__ h_out_bf,
                                                       float* __restrict__ out,
                                                       int t) {
  using namespace gru_cfg;
  __shared__ float ghs[3 * 32 * 16];   // [gate][row-in-block][col] = 6 KB

  const int tid   = threadIdx.x;
  const int lane  = tid & 31;
  const int w     = tid >> 5;     // 0..5
  const int g     = w >> 1;       // gate 0..2
  const int mt    = w & 1;        // row tile within block
  const int j0    = blockIdx.x * 16;
  const int rowb  = blockIdx.y * 32;           // 0 or 32
  const int r16   = lane & 15;
  const int khalf = lane >> 4;

  const __bf16* aptr = h_in_bf + (long)(rowb + mt * 16 + r16) * H;
  const __bf16* bptr = whh + (long)(g * H + j0 + r16) * H;

  v8f c = {};
#pragma unroll 4
  for (int k0 = 0; k0 < H; k0 += 32) {
    const int ka = k0 + khalf * 8;
    v8bf alo = *(const v8bf*)(aptr + ka);
    v8bf ahi = *(const v8bf*)(aptr + ka + 16);
    v8bf blo = *(const v8bf*)(bptr + ka);
    v8bf bhi = *(const v8bf*)(bptr + ka + 16);
    v16bf a = CAT16(alo, ahi);
    v16bf b = CAT16(blo, bhi);
    c = __builtin_amdgcn_wmma_f32_16x16x32_bf16(false, a, false, b, (short)0, c, false, false);
  }

  // scatter gh tile to LDS: rows khalf*8..+7 of col r16
  {
    float* dst = &ghs[(g * 32 + mt * 16 + khalf * 8) * 16 + r16];
#pragma unroll
    for (int i = 0; i < 8; ++i) dst[i * 16] = c[i];
  }
  __syncthreads();

  // gate math: 32x16 = 512 outputs over 192 threads
  for (int idx = tid; idx < 32 * 16; idx += 192) {
    const int brow = idx >> 4;               // 0..31 within block
    const int b    = rowb + brow;            // batch row
    const int jl   = idx & 15;
    const int j    = j0 + jl;

    const float hr = ghs[(0 * 32 + brow) * 16 + jl] + bhh[j];
    const float hi = ghs[(1 * 32 + brow) * 16 + jl] + bhh[H + j];
    const float hn = ghs[(2 * 32 + brow) * 16 + jl] + bhh[2 * H + j];

    const float* gib = gi + ((long)b * T + t) * (long)G3;
    const float ir  = gib[j];
    const float ii  = gib[H + j];
    const float in_ = gib[2 * H + j];

    const float r = 1.0f / (1.0f + __expf(-(ir + hr)));
    const float z = 1.0f / (1.0f + __expf(-(ii + hi)));
    const float n = tanhf(in_ + r * hn);
    const float hp = h_in[(long)b * H + j];
    const float hy = n + z * (hp - n);

    out[((long)b * T + t) * (long)H + j] = hy;
    h_out[(long)b * H + j]    = hy;
    h_out_bf[(long)b * H + j] = (__bf16)hy;
  }
}

// ---------------------------------------------------------------------------
// Launch
// ---------------------------------------------------------------------------
extern "C" void kernel_launch(void* const* d_in, const int* in_sizes, int n_in,
                              void* d_out, int out_size, void* d_ws, size_t ws_size,
                              hipStream_t stream) {
  using namespace gru_cfg;
  const float* x   = (const float*)d_in[0];   // [B,T,I]
  const float* wih = (const float*)d_in[1];   // [3H,I]
  const float* whh = (const float*)d_in[2];   // [3H,H]
  const float* bih = (const float*)d_in[3];   // [3H]
  const float* bhh = (const float*)d_in[4];   // [3H]
  float* out = (float*)d_out;                 // [B,T,H] ++ [B,H]

  // workspace carve-up (all offsets 16B aligned)
  char* p = (char*)d_ws;
  __bf16* wih_bf = (__bf16*)p;  p += (size_t)G3 * I * sizeof(__bf16);   // 6.29 MB
  __bf16* whh_bf = (__bf16*)p;  p += (size_t)G3 * H * sizeof(__bf16);   // 6.29 MB
  float*  hf0    = (float*)p;   p += (size_t)B * H * sizeof(float);
  float*  hf1    = (float*)p;   p += (size_t)B * H * sizeof(float);
  __bf16* hb0    = (__bf16*)p;  p += (size_t)B * H * sizeof(__bf16);
  __bf16* hb1    = (__bf16*)p;  p += (size_t)B * H * sizeof(__bf16);
  float*  gi     = (float*)p;   p += (size_t)BT * G3 * sizeof(float);   // 402.7 MB
  float*  hf[2]  = {hf0, hf1};
  __bf16* hb[2]  = {hb0, hb1};

  // 1) weights -> bf16 (L2-resident for the recurrence)
  f32_to_bf16_kernel<<<(G3 * I) / 256, 256, 0, stream>>>(wih, wih_bf, G3 * I);
  f32_to_bf16_kernel<<<(G3 * H) / 256, 256, 0, stream>>>(whh, whh_bf, G3 * H);
  // 2) h0 = 0 (both fp32 and bf16 mirrors)
  zero_h_kernel<<<(B * H) / 256, 256, 0, stream>>>(hf[0], hb[0], B * H);
  // 3) input projection GEMM: [32768 x 3072 x 1024] bf16 WMMA, 2x2 reg-blocked
  gi_gemm_kernel<<<dim3(G3 / 64, BT / 128), 256, 0, stream>>>(x, wih_bf, bih, gi);
  // 4) 512 fused recurrence steps (double-buffered h to avoid RAW races)
  for (int t = 0; t < T; ++t) {
    const int in = t & 1, o = in ^ 1;
    gru_step_kernel<<<dim3(H / 16, 2), 192, 0, stream>>>(
        gi, whh_bf, bhh, hf[in], hb[in], hf[o], hb[o], out, t);
  }
  // 5) h_last (after t=T-1, result lives in buffer 0) -> output tail
  hipMemcpyAsync(out + (size_t)BT * H, hf[0], (size_t)B * H * sizeof(float),
                 hipMemcpyDeviceToDevice, stream);
}
